// RegistFormer_3676492005678
// MI455X (gfx1250) — matrix operand order repaired and checked by reading, not compiled, not run on hardware
//
#include <hip/hip_runtime.h>
#include <hip/hip_bf16.h>
#include <math.h>

typedef _Float16 v16h __attribute__((ext_vector_type(16)));
typedef float    v8f  __attribute__((ext_vector_type(8)));
typedef float    v4f  __attribute__((ext_vector_type(4)));
typedef unsigned int u32x4 __attribute__((ext_vector_type(4)));
typedef int          i32x8 __attribute__((ext_vector_type(8)));
typedef int          i32x4 __attribute__((ext_vector_type(4)));

#define HWN   16384      // H*W
#define Hd    128
#define Wd    128
#define Cdim  128
#define GN_EPS 1e-5f

#if __has_builtin(__builtin_amdgcn_tensor_load_to_lds)
#define HAVE_TDM 1
#else
#define HAVE_TDM 0
#endif

#if HAVE_TDM
// Issue a TDM 2D tile load: tileN (contig, data_size=4B) x tileK rows with
// row stride strideN (elements) from gsrc into LDS byte offset lds_byte.
// D# built per CDNA5 ISA 8.3/8.4 (group0: count/lds/global/type=2;
// group1: data_size=4B, tensor_dim0/1 == tile_dim0/1 -> never OOB).
// This toolchain exposes the 6-arg builtin form:
//   (uint32x4 g0, int32x8 g1, int32x4 g2, int32x4 g3, int32x8 extra, i32 cpol)
__device__ __forceinline__ void tdm_issue_tile(const float* gsrc, unsigned lds_byte,
                                               int tileN, int tileK, int strideN)
{
    unsigned long long ga = (unsigned long long)(uintptr_t)gsrc;
    u32x4 g0;
    g0[0] = 1u;                                   // count=1 (valid), user mode
    g0[1] = lds_byte;                             // lds_addr (bytes)
    g0[2] = (unsigned)(ga & 0xFFFFFFFFu);         // global_addr[31:0]  (bits 95:64)
    g0[3] = (unsigned)(ga >> 32) | (2u << 30);    // global_addr[56:32] | type=2
    i32x8 g1;
    g1[0] = (2 << 16);                            // data_size=2 -> 4 bytes
    g1[1] = (tileN & 0xFFFF) << 16;               // tensor_dim0 lo16 (bits 63:48)
    g1[2] = ((tileN >> 16) & 0xFFFF) | ((tileK & 0xFFFF) << 16); // dim0 hi | dim1 lo
    g1[3] = ((tileK >> 16) & 0xFFFF) | ((tileN & 0xFFFF) << 16); // dim1 hi | tile_dim0
    g1[4] = (tileK & 0xFFFF);                     // tile_dim1 (tile_dim2=0)
    g1[5] = strideN;                              // tensor_dim0_stride lo32
    g1[6] = 0;                                    // stride0 hi | stride1 lo
    g1[7] = 0;
    i32x4 z4 = {0, 0, 0, 0};
    i32x8 z8 = {0, 0, 0, 0, 0, 0, 0, 0};
    __builtin_amdgcn_tensor_load_to_lds(g0, g1, z4, z4, z8, 0);
}
#endif

// ---------------------------------------------------------------------------
// GEMM: D[M,N=16384] = W[M,K] x X[K,N]  (+bias, leaky, +residual), via WMMA.
// Block = 256 threads = 8 waves; block tile 128(M) x 32(N); each wave computes
// 16x32 (2 accumulators, A reused). X tile (32K x 32N, f32) staged to LDS by
// the Tensor Data Mover (double-buffered) when available.
// ---------------------------------------------------------------------------
__global__ void __launch_bounds__(256)
gemm_wmma_kernel(const float* __restrict__ Wt, const float* __restrict__ X,
                 float* __restrict__ D, int M, int K,
                 const float* __restrict__ bias, const float* __restrict__ resid,
                 int leaky)
{
    const int N = HWN;
    const int wave = threadIdx.x >> 5;
    const int lane = threadIdx.x & 31;
    const int g    = lane >> 4;          // lane group (0/1)
    const int ln16 = lane & 15;
    const int p0 = blockIdx.x * 32;      // N offset (32-wide block tile)
    const int m0 = blockIdx.y * 128 + wave * 16;
    const int row = m0 + ln16;           // A-matrix M row for this lane

    __shared__ __align__(16) float xt[2][32 * 32];   // [buf][k][n] f32

    v8f acc0 = {0.f,0.f,0.f,0.f,0.f,0.f,0.f,0.f};
    v8f acc1 = {0.f,0.f,0.f,0.f,0.f,0.f,0.f,0.f};

    const int iters = K >> 5;

#if HAVE_TDM
    const bool issuer = (wave == 0);
    if (issuer)
        tdm_issue_tile(X + p0, (unsigned)(uintptr_t)&xt[0][0], 32, 32, N);
#endif

    for (int it = 0; it < iters; ++it) {
        const int k0 = it << 5;
#if HAVE_TDM
        if (issuer) {
            if (it + 1 < iters) {
                tdm_issue_tile(X + (size_t)(k0 + 32) * N + p0,
                               (unsigned)(uintptr_t)&xt[(it + 1) & 1][0], 32, 32, N);
                __builtin_amdgcn_s_wait_tensorcnt(1);   // tile `it` landed
            } else {
                __builtin_amdgcn_s_wait_tensorcnt(0);
            }
        }
        __syncthreads();                                // tile `it` visible
        const float* bt = &xt[it & 1][0];
#else
        for (int i = threadIdx.x; i < 1024; i += 256) {
            int kr = i >> 5, nn = i & 31;
            xt[0][kr * 32 + nn] = X[(size_t)(k0 + kr) * N + p0 + nn];
        }
        if (it + 1 < iters)
            __builtin_prefetch(&X[(size_t)(k0 + 32 + (threadIdx.x >> 5)) * N
                                  + p0 + (threadIdx.x & 31)], 0, 1);
        __syncthreads();
        const float* bt = &xt[0][0];
#endif

        // ---- A operand: W[row, k0 + {g*8..+7, 16+g*8..+7}] (16-bit A layout)
        const float* wr = Wt + (size_t)row * K + k0;
        v4f a0 = *(const v4f*)(wr + 8 * g);
        v4f a1 = *(const v4f*)(wr + 8 * g + 4);
        v4f a2 = *(const v4f*)(wr + 16 + 8 * g);
        v4f a3 = *(const v4f*)(wr + 16 + 8 * g + 4);
        v16h a;
        #pragma unroll
        for (int i = 0; i < 4; ++i) {
            a[i]      = (_Float16)a0[i];
            a[4 + i]  = (_Float16)a1[i];
            a[8 + i]  = (_Float16)a2[i];
            a[12 + i] = (_Float16)a3[i];
        }

        // ---- B operands from LDS [k][n]: lane n = ln16(+16), K = 8*g + e (+8 for e >= 8) --
        v16h b0v, b1v;
        #pragma unroll
        for (int e = 0; e < 16; ++e) {
            int kk = (8 * g + e + (e >= 8 ? 8 : 0)) * 32;
            b0v[e] = (_Float16)bt[kk + ln16];
            b1v[e] = (_Float16)bt[kk + ln16 + 16];
        }

        acc0 = __builtin_amdgcn_wmma_f32_16x16x32_f16(
                   false, a, false, b0v, (short)0, acc0, false, false);
        acc1 = __builtin_amdgcn_wmma_f32_16x16x32_f16(
                   false, a, false, b1v, (short)0, acc1, false, false);
        __syncthreads();                                // done reading tile
    }

    // ---- epilogue: D layout lane=(n), VGPR j -> M = j + 8*g ----
    const int col = p0 + ln16;
    #pragma unroll
    for (int j = 0; j < 8; ++j) {
        int rowo = m0 + j + 8 * g;
        float v0 = acc0[j];
        float v1 = acc1[j];
        if (bias) { float b = bias[rowo]; v0 += b; v1 += b; }
        if (leaky) {
            v0 = (v0 >= 0.f) ? v0 : 0.2f * v0;
            v1 = (v1 >= 0.f) ? v1 : 0.2f * v1;
        }
        if (resid) {
            v0 += resid[(size_t)rowo * N + col];
            v1 += resid[(size_t)rowo * N + col + 16];
        }
        D[(size_t)rowo * N + col]      = v0;
        D[(size_t)rowo * N + col + 16] = v1;
    }
}

// ---------------------------------------------------------------------------
// Deformable attention. One wave = 32 consecutive pixels (same row) for one
// head. All 25 samples share one fractional offset -> gather a 6x6 patch per
// channel, bilinear-combine with 4 constant weights. In-lane softmax over 25.
// ---------------------------------------------------------------------------
__global__ void __launch_bounds__(256)
deform_attn_kernel(const float* __restrict__ q, const float* __restrict__ k,
                   const float* __restrict__ v, const float* __restrict__ df,
                   float* __restrict__ out)
{
    const int head = threadIdx.x >> 5;           // 8 waves = 8 heads
    const int lane = threadIdx.x & 31;
    const int p    = blockIdx.x * 32 + lane;     // pixel id
    const int px   = p & (Wd - 1);
    const int py   = p >> 7;

    const float dx = df[p];
    const float dy = df[HWN + p];
    const float fx = (float)px + dx;
    const float fy = (float)py + dy;
    const float x0f = floorf(fx), y0f = floorf(fy);
    const float wx1 = fx - x0f, wy1 = fy - y0f;
    const float wx0 = 1.f - wx1, wy0 = 1.f - wy1;
    const int xb = (int)x0f, yb = (int)y0f;
    const float scale = 0.25f;                   // 1/sqrt(d_k=16)

    float scores[25];
    #pragma unroll
    for (int s = 0; s < 25; ++s) scores[s] = 0.f;

    // ---------------- phase 1: scores over this head's 16 channels --------
    for (int c = 0; c < 16; ++c) {
        const int cc = head * 16 + c;
        const float qv = q[(size_t)cc * HWN + p] * scale;
        const float* kc = k + (size_t)cc * HWN;
        float patch[6][6];
        #pragma unroll
        for (int iy = 0; iy < 6; ++iy) {
            int yy = yb - 2 + iy;
            bool oky = (yy >= 0) && (yy < Hd);
            int yc = yy < 0 ? 0 : (yy > Hd - 1 ? Hd - 1 : yy);
            #pragma unroll
            for (int ix = 0; ix < 6; ++ix) {
                int xx = xb - 2 + ix;
                bool ok = oky && (xx >= 0) && (xx < Wd);
                int xc = xx < 0 ? 0 : (xx > Wd - 1 ? Wd - 1 : xx);
                float val = kc[yc * Wd + xc];
                patch[iy][ix] = ok ? val : 0.f;
            }
        }
        #pragma unroll
        for (int s = 0; s < 25; ++s) {
            int iy = s / 5, ix = s % 5;
            float kk = wy0 * (wx0 * patch[iy][ix]     + wx1 * patch[iy][ix + 1])
                     + wy1 * (wx0 * patch[iy + 1][ix] + wx1 * patch[iy + 1][ix + 1]);
            scores[s] += qv * kk;
        }
    }

    // ---------------- softmax over 25 samples (in-lane) --------------------
    float m = -3.4e38f;
    #pragma unroll
    for (int s = 0; s < 25; ++s) m = fmaxf(m, scores[s]);
    float ssum = 0.f;
    #pragma unroll
    for (int s = 0; s < 25; ++s) { scores[s] = __expf(scores[s] - m); ssum += scores[s]; }
    float inv = 1.f / ssum;
    #pragma unroll
    for (int s = 0; s < 25; ++s) scores[s] *= inv;

    // ---------------- phase 2: out = sum_s attn[s] * v_s -------------------
    for (int c = 0; c < 16; ++c) {
        const int cc = head * 16 + c;
        const float* vc = v + (size_t)cc * HWN;
        float patch[6][6];
        #pragma unroll
        for (int iy = 0; iy < 6; ++iy) {
            int yy = yb - 2 + iy;
            bool oky = (yy >= 0) && (yy < Hd);
            int yc = yy < 0 ? 0 : (yy > Hd - 1 ? Hd - 1 : yy);
            #pragma unroll
            for (int ix = 0; ix < 6; ++ix) {
                int xx = xb - 2 + ix;
                bool ok = oky && (xx >= 0) && (xx < Wd);
                int xc = xx < 0 ? 0 : (xx > Wd - 1 ? Wd - 1 : xx);
                float val = vc[yc * Wd + xc];
                patch[iy][ix] = ok ? val : 0.f;
            }
        }
        float accv = 0.f;
        #pragma unroll
        for (int s = 0; s < 25; ++s) {
            int iy = s / 5, ix = s % 5;
            float vv = wy0 * (wx0 * patch[iy][ix]     + wx1 * patch[iy][ix + 1])
                     + wy1 * (wx0 * patch[iy + 1][ix] + wx1 * patch[iy + 1][ix + 1]);
            accv += scores[s] * vv;
        }
        out[(size_t)cc * HWN + p] = accv;
    }
}

// ---------------------------------------------------------------------------
// Layer-norm over whole tensor (mean/var over C*H*W), two-stage deterministic.
// ---------------------------------------------------------------------------
__global__ void __launch_bounds__(256)
reduce_partial_kernel(const float* __restrict__ x, float* __restrict__ part, int n)
{
    __shared__ float ssum[256], ssq[256];
    float s = 0.f, sq = 0.f;
    for (int i = blockIdx.x * 256 + threadIdx.x; i < n; i += 256 * 256) {
        float val = x[i];
        s += val; sq += val * val;
    }
    ssum[threadIdx.x] = s; ssq[threadIdx.x] = sq;
    __syncthreads();
    for (int off = 128; off > 0; off >>= 1) {
        if (threadIdx.x < off) {
            ssum[threadIdx.x] += ssum[threadIdx.x + off];
            ssq[threadIdx.x]  += ssq[threadIdx.x + off];
        }
        __syncthreads();
    }
    if (threadIdx.x == 0) {
        part[blockIdx.x]       = ssum[0];
        part[256 + blockIdx.x] = ssq[0];
    }
}

__global__ void __launch_bounds__(256)
reduce_final_kernel(const float* __restrict__ part, float* __restrict__ mv, int n)
{
    __shared__ float ssum[256], ssq[256];
    ssum[threadIdx.x] = part[threadIdx.x];
    ssq[threadIdx.x]  = part[256 + threadIdx.x];
    __syncthreads();
    for (int off = 128; off > 0; off >>= 1) {
        if (threadIdx.x < off) {
            ssum[threadIdx.x] += ssum[threadIdx.x + off];
            ssq[threadIdx.x]  += ssq[threadIdx.x + off];
        }
        __syncthreads();
    }
    if (threadIdx.x == 0) {
        float mean = ssum[0] / (float)n;
        float var  = ssq[0] / (float)n - mean * mean;
        mv[0] = mean;
        mv[1] = rsqrtf(var + GN_EPS);
    }
}

__global__ void __launch_bounds__(256)
norm_kernel(float* __restrict__ out, const float* __restrict__ mv,
            const float* __restrict__ gw, const float* __restrict__ gb)
{
    int i = blockIdx.x * 256 + threadIdx.x;
    int c = i >> 14;                      // / HWN
    float val = out[i];
    out[i] = (val - mv[0]) * mv[1] * gw[c] + gb[c];
}

// ---------------------------------------------------------------------------
extern "C" void kernel_launch(void* const* d_in, const int* in_sizes, int n_in,
                              void* d_out, int out_size, void* d_ws, size_t ws_size,
                              hipStream_t stream)
{
    const float* query = (const float*)d_in[0];
    const float* key   = (const float*)d_in[1];
    const float* value = (const float*)d_in[2];
    const float* df    = (const float*)d_in[3];
    const float* w_q   = (const float*)d_in[4];
    const float* w_k   = (const float*)d_in[5];
    const float* w_v   = (const float*)d_in[6];
    const float* w_fc  = (const float*)d_in[7];
    const float* mw1   = (const float*)d_in[8];
    const float* mb1   = (const float*)d_in[9];
    const float* mw2   = (const float*)d_in[10];
    const float* mb2   = (const float*)d_in[11];
    const float* gnw   = (const float*)d_in[12];
    const float* gnb   = (const float*)d_in[13];
    float* out = (float*)d_out;
    float* ws  = (float*)d_ws;

    const size_t CHW = (size_t)Cdim * HWN;       // 2,097,152
    float* q_buf   = ws;                          // [0, CHW)
    float* k_buf   = ws + CHW;                    // [CHW, 2CHW)
    float* v_buf   = ws + 2 * CHW;                // [2CHW, 3CHW)
    float* attn    = ws + 3 * CHW;                // [3CHW, 4CHW)
    float* fc_buf  = ws;                          // reuse q slot
    float* hid     = ws + CHW;                    // reuse k+v slots (2*CHW floats)
    float* part    = ws + 4 * CHW;                // 512 floats
    float* mv      = ws + 4 * CHW + 512;          // 2 floats

    dim3 blk(256);
    dim3 gemm128(HWN / 32, 1);
    dim3 gemm256(HWN / 32, 2);

    // q,k,v projections
    gemm_wmma_kernel<<<gemm128, blk, 0, stream>>>(w_q, query, q_buf, 128, 128, nullptr, nullptr, 0);
    gemm_wmma_kernel<<<gemm128, blk, 0, stream>>>(w_k, key,   k_buf, 128, 128, nullptr, nullptr, 0);
    gemm_wmma_kernel<<<gemm128, blk, 0, stream>>>(w_v, value, v_buf, 128, 128, nullptr, nullptr, 0);

    // deformable attention
    deform_attn_kernel<<<dim3(HWN / 32), blk, 0, stream>>>(q_buf, k_buf, v_buf, df, attn);

    // fc projection
    gemm_wmma_kernel<<<gemm128, blk, 0, stream>>>(w_fc, attn, fc_buf, 128, 128, nullptr, nullptr, 0);

    // MLP: hid = leaky(w1 x fc + b1); out = fc + (w2 x hid + b2)
    gemm_wmma_kernel<<<gemm256, blk, 0, stream>>>(mw1, fc_buf, hid, 256, 128, mb1, nullptr, 1);
    gemm_wmma_kernel<<<gemm128, blk, 0, stream>>>(mw2, hid, out, 128, 256, mb2, fc_buf, 0);

    // layer norm (in place in d_out)
    reduce_partial_kernel<<<dim3(256), blk, 0, stream>>>(out, part, (int)CHW);
    reduce_final_kernel<<<dim3(1), blk, 0, stream>>>(part, mv, (int)CHW);
    norm_kernel<<<dim3((int)(CHW / 256)), blk, 0, stream>>>(out, mv, gnw, gnb);
}
